// TransformerBlock_46806553591881
// MI455X (gfx1250) — compile-verified, matrix operand order checked by reading, not run on hardware
//
#include <hip/hip_runtime.h>
#include <hip/hip_bf16.h>
#include <cstdint>

typedef __bf16 bf16_t;
typedef __bf16 v16bf __attribute__((ext_vector_type(16)));
typedef __bf16 v8bf  __attribute__((ext_vector_type(8)));
typedef float  v8f   __attribute__((ext_vector_type(8)));
typedef uint32_t u32x4 __attribute__((ext_vector_type(4)));
typedef uint32_t u32x8 __attribute__((ext_vector_type(8)));

union V16 { v16bf v; v8bf h[2]; };

constexpr int BB = 2, SS = 2048, DD = 1024, HH = 16, HD = 64, FF = 4096;
constexpr int RR = BB * SS;   // 4096 total rows

__device__ __forceinline__ v8f wmma_bf16(v16bf a, v16bf b, v8f c) {
  // D = A(16x32 bf16) * B(32x16 bf16) + C(16x16 f32)
  return __builtin_amdgcn_wmma_f32_16x16x32_bf16(false, a, false, b, (short)0, c,
                                                 false, false);
}

// ---------------------------------------------------------------------------
// TDM: stage a [rows=128] x [cols=32] bf16 tile (row stride = K elements in
// global) into LDS at lds_off, rows stored back-to-back (64B each).
// D# layout per CDNA5 ISA ch.8 (async_tensor): group0 = count|lds|gaddr|type,
// group1 = data_size, tensor dims/strides, tile dims. Tracked by TENSORcnt.
// ---------------------------------------------------------------------------
__device__ __forceinline__ void tdm_load_tile(uint32_t lds_off, uint64_t gaddr,
                                              uint32_t dim0, uint32_t dim1) {
  u32x4 g0;
  g0.x = 1u;                                                    // count=1 (valid), user mode
  g0.y = lds_off;                                               // lds_addr[31:0]
  g0.z = (uint32_t)gaddr;                                       // global_addr[31:0]
  g0.w = (uint32_t)((gaddr >> 32) & 0x01FFFFFFu) | (2u << 30);  // addr[56:32] | type=2

  u32x8 g1;
  g1.s0 = 1u << 16;                              // workgroup_mask=0, data_size=1 (2B)
  g1.s1 = (dim0 & 0xFFFFu) << 16;                // tensor_dim0[15:0]  (bits 63:48)
  g1.s2 = (dim0 >> 16) | ((dim1 & 0xFFFFu) << 16); // tensor_dim0[31:16] | tensor_dim1[15:0]
  g1.s3 = (dim1 >> 16) | (32u << 16);            // tensor_dim1[31:16] | tile_dim0=32
  g1.s4 = 128u;                                  // tile_dim1=128, tile_dim2=0
  g1.s5 = dim0;                                  // tensor_dim0_stride[31:0] = K
  g1.s6 = 0u;                                    // stride0[47:32] | stride1[15:0]
  g1.s7 = 0u;                                    // stride1[47:16]
  asm volatile("tensor_load_to_lds %0, %1" :: "s"(g0), "s"(g1) : "memory");
}

// ---------------------------------------------------------------------------
// Weight convert + transpose: w[K,N] f32 -> wT[N,K] bf16
// ---------------------------------------------------------------------------
__global__ void wconv_kernel(const float* __restrict__ w, bf16_t* __restrict__ wt,
                             int K, int N) {
  size_t idx = (size_t)blockIdx.x * blockDim.x + threadIdx.x;
  if (idx >= (size_t)K * N) return;
  int k = (int)(idx / N), n = (int)(idx % N);
  wt[(size_t)n * K + k] = (bf16_t)w[idx];
}

// ---------------------------------------------------------------------------
// LayerNorm over D=1024, one block (256 thr) per row, bf16 output
// ---------------------------------------------------------------------------
__global__ __launch_bounds__(256) void ln_kernel(const float* __restrict__ x,
                                                 const float* __restrict__ g,
                                                 const float* __restrict__ bta,
                                                 bf16_t* __restrict__ out) {
  const int row = blockIdx.x, t = threadIdx.x;
  const float4* xr = (const float4*)(x + (size_t)row * DD);
  float4 v = xr[t];
  float s = v.x + v.y + v.z + v.w;
  float q = v.x * v.x + v.y * v.y + v.z * v.z + v.w * v.w;
  __shared__ float s1[256], s2[256];
  s1[t] = s; s2[t] = q; __syncthreads();
  for (int st = 128; st > 0; st >>= 1) {
    if (t < st) { s1[t] += s1[t + st]; s2[t] += s2[t + st]; }
    __syncthreads();
  }
  float mu  = s1[0] * (1.0f / DD);
  float var = s2[0] * (1.0f / DD) - mu * mu;
  float rs  = rsqrtf(var + 1e-5f);
  bf16_t* orow = out + (size_t)row * DD;
#pragma unroll
  for (int i = 0; i < 4; ++i) {
    int c = t * 4 + i;
    float xv = (&v.x)[i];
    orow[c] = (bf16_t)((xv - mu) * rs * g[c] + bta[c]);
  }
}

// ---------------------------------------------------------------------------
// bf16 GEMM: C[M,N] = A[M,K] * Bt[N,K]^T  (Bt pre-transposed)
// 256 threads = 8 waves; block tile 128x128; wave tile 32x64 (2x4 WMMA tiles).
// A/B K-slices staged to LDS by the Tensor Data Mover, double buffered;
// wave 0 issues the DMA one K-step ahead and gates with s_wait_tensorcnt.
// EPI: 0 -> store bf16 ; 1 -> f32 + bias + residual ; 2 -> bf16 gelu(bias+v)
// ---------------------------------------------------------------------------
template <int EPI>
__global__ __launch_bounds__(256) void gemm_bf16_kernel(
    const bf16_t* __restrict__ A, const bf16_t* __restrict__ Bt,
    int M, int N, int K,
    const float* __restrict__ bias, const float* __restrict__ resid,
    float* __restrict__ outF, bf16_t* __restrict__ outB) {
  __shared__ __align__(16) bf16_t ldsA[2][128][32];   // 2 x 8KB
  __shared__ __align__(16) bf16_t ldsB[2][128][32];   // 2 x 8KB

  const int wave = threadIdx.x >> 5;
  const int lane = threadIdx.x & 31;
  const int half = lane >> 4, lr = lane & 15;
  const int wml = (wave & 3) * 32;   // 4 waves tile M
  const int wnl = (wave >> 2) * 64;  // 2 waves tile N
  const int bm = blockIdx.x * 128, bn = blockIdx.y * 128;

  const uint64_t aBase = (uint64_t)(uintptr_t)A + (uint64_t)bm * K * 2u;
  const uint64_t bBase = (uint64_t)(uintptr_t)Bt + (uint64_t)bn * K * 2u;
  // Flat shared addresses are {aperture[63:32], lds_offset[31:0]} (ISA 10.2):
  const uint32_t ldsAoff = (uint32_t)(uintptr_t)&ldsA[0][0][0];
  const uint32_t ldsBoff = (uint32_t)(uintptr_t)&ldsB[0][0][0];
  constexpr uint32_t BUFB = 128u * 32u * 2u;   // 8KB per buffer

  if (wave == 0) {  // prologue: stage k0=0 into buffer 0 (EXEC ignored by TDM)
    tdm_load_tile(ldsAoff, aBase, (uint32_t)K, (uint32_t)M);
    tdm_load_tile(ldsBoff, bBase, (uint32_t)K, (uint32_t)N);
  }

  v8f c[2][4] = {};
  int cur = 0;
  for (int k0 = 0; k0 < K; k0 += 32, cur ^= 1) {
    if (wave == 0) {
      if (k0 + 32 < K) {   // stage next K-slice into the other buffer
        const uint32_t nxt = (uint32_t)(cur ^ 1) * BUFB;
        tdm_load_tile(ldsAoff + nxt, aBase + (uint64_t)(k0 + 32) * 2u,
                      (uint32_t)K, (uint32_t)M);
        tdm_load_tile(ldsBoff + nxt, bBase + (uint64_t)(k0 + 32) * 2u,
                      (uint32_t)K, (uint32_t)N);
        __builtin_amdgcn_s_wait_tensorcnt(2);  // in-order: current tile resident
      } else {
        __builtin_amdgcn_s_wait_tensorcnt(0);
      }
    }
    __syncthreads();

    V16 a[2], b[4];
#pragma unroll
    for (int i = 0; i < 2; ++i) {
      // 16-bit A/B fragment: row/col = lr; lanes<16 hold K 0..7 & 16..23,
      // lanes>=16 hold K 8..15 & 24..31 (ISA 7.12.2)
      a[i].h[0] = *(const v8bf*)&ldsA[cur][wml + i * 16 + lr][half * 8];
      a[i].h[1] = *(const v8bf*)&ldsA[cur][wml + i * 16 + lr][16 + half * 8];
    }
#pragma unroll
    for (int j = 0; j < 4; ++j) {
      b[j].h[0] = *(const v8bf*)&ldsB[cur][wnl + j * 16 + lr][half * 8];
      b[j].h[1] = *(const v8bf*)&ldsB[cur][wnl + j * 16 + lr][16 + half * 8];
    }
#pragma unroll
    for (int i = 0; i < 2; ++i)
#pragma unroll
      for (int j = 0; j < 4; ++j)
        c[i][j] = wmma_bf16(a[i].v, b[j].v, c[i][j]);
    __syncthreads();   // readers done before TDM overwrites this buffer
  }

#pragma unroll
  for (int i = 0; i < 2; ++i) {
#pragma unroll
    for (int j = 0; j < 4; ++j) {
#pragma unroll
      for (int r = 0; r < 8; ++r) {
        // C layout: VGPR r, lanes<16 -> M=r, lanes>=16 -> M=8+r, N=lr
        int row = bm + wml + i * 16 + half * 8 + r;
        int col = bn + wnl + j * 16 + lr;
        size_t idx = (size_t)row * N + col;
        float v = c[i][j][r];
        if constexpr (EPI == 0) {
          outB[idx] = (bf16_t)v;
        } else if constexpr (EPI == 1) {
          outF[idx] = v + bias[col] + resid[idx];
        } else {
          float t = v + bias[col];
          outB[idx] = (bf16_t)(0.5f * t * (1.0f + erff(t * 0.70710678118654752f)));
        }
      }
    }
  }
}

// ---------------------------------------------------------------------------
// Split heads: qb/kb/vb [B*S, H*hd] -> qh/kh [B,H,S,hd], vT [B,H,hd,S]
// ---------------------------------------------------------------------------
__global__ void split_heads_kernel(const bf16_t* __restrict__ qb,
                                   const bf16_t* __restrict__ kb,
                                   const bf16_t* __restrict__ vb,
                                   bf16_t* __restrict__ qh, bf16_t* __restrict__ kh,
                                   bf16_t* __restrict__ vT) {
  size_t e = (size_t)blockIdx.x * blockDim.x + threadIdx.x;
  if (e >= (size_t)RR * DD) return;
  int sg = (int)(e / DD), cc = (int)(e % DD);
  int h = cc / HD, c = cc % HD;
  int b = sg / SS, s = sg % SS;
  size_t bh = (size_t)b * HH + h;
  qh[(bh * SS + s) * HD + c] = qb[e];
  kh[(bh * SS + s) * HD + c] = kb[e];
  vT[(bh * HD + c) * SS + s] = vb[e];
}

// ---------------------------------------------------------------------------
// Flash-style attention. Grid = B*H*(S/64); 128 threads = 4 waves; each wave
// owns 16 query rows. Online softmax in WMMA C-layout; P transposed to
// A-layout through a 1KB/wave LDS bounce.
// ---------------------------------------------------------------------------
__global__ __launch_bounds__(128) void attn_kernel(const bf16_t* __restrict__ qh,
                                                   const bf16_t* __restrict__ kh,
                                                   const bf16_t* __restrict__ vT,
                                                   bf16_t* __restrict__ attB) {
  const int bh = blockIdx.x / (SS / 64);
  const int qt = blockIdx.x % (SS / 64);
  const int b = bh / HH, h = bh % HH;
  const int wave = threadIdx.x >> 5, lane = threadIdx.x & 31;
  const int half = lane >> 4, lr = lane & 15;
  const int qrow0 = qt * 64 + wave * 16;

  const bf16_t* Q  = qh + (size_t)bh * SS * HD;
  const bf16_t* Kp = kh + (size_t)bh * SS * HD;
  const bf16_t* Vp = vT + (size_t)bh * HD * SS;

  __shared__ __align__(16) bf16_t pbuf[4][16][32];

  V16 aq[2];
  {
    const bf16_t* qr = Q + (size_t)(qrow0 + lr) * HD;
#pragma unroll
    for (int kk = 0; kk < 2; ++kk) {
      aq[kk].h[0] = *(const v8bf*)(qr + kk * 32 + half * 8);
      aq[kk].h[1] = *(const v8bf*)(qr + kk * 32 + 16 + half * 8);
    }
  }

  v8f O[4] = {};
  float mrow[8], lrow[8];
#pragma unroll
  for (int r = 0; r < 8; ++r) { mrow[r] = -1e30f; lrow[r] = 0.0f; }

  for (int kt = 0; kt < SS; kt += 32) {
    v8f sc[2];
#pragma unroll
    for (int j = 0; j < 2; ++j) {
      const bf16_t* kr = Kp + (size_t)(kt + j * 16 + lr) * HD;
      V16 b0, b1;
      b0.h[0] = *(const v8bf*)(kr + half * 8);
      b0.h[1] = *(const v8bf*)(kr + 16 + half * 8);
      b1.h[0] = *(const v8bf*)(kr + 32 + half * 8);
      b1.h[1] = *(const v8bf*)(kr + 48 + half * 8);
      v8f z = {};
      v8f t = wmma_bf16(aq[0].v, b0.v, z);
      t = wmma_bf16(aq[1].v, b1.v, t);
      sc[j] = t * 0.125f;  // 1/sqrt(hd)
    }

    float alpha[8];
#pragma unroll
    for (int r = 0; r < 8; ++r) {
      float mx = fmaxf(sc[0][r], sc[1][r]);
#pragma unroll
      for (int off = 8; off >= 1; off >>= 1) mx = fmaxf(mx, __shfl_xor(mx, off));
      float mnew = fmaxf(mrow[r], mx);
      alpha[r] = __expf(mrow[r] - mnew);
      mrow[r] = mnew;
      float p0 = __expf(sc[0][r] - mnew);
      float p1 = __expf(sc[1][r] - mnew);
      sc[0][r] = p0; sc[1][r] = p1;
      float sum = p0 + p1;
#pragma unroll
      for (int off = 8; off >= 1; off >>= 1) sum += __shfl_xor(sum, off);
      lrow[r] = lrow[r] * alpha[r] + sum;
    }
#pragma unroll
    for (int nt = 0; nt < 4; ++nt)
#pragma unroll
      for (int r = 0; r < 8; ++r) O[nt][r] *= alpha[r];

    __syncthreads();
#pragma unroll
    for (int r = 0; r < 8; ++r) {
      pbuf[wave][half * 8 + r][lr]      = (bf16_t)sc[0][r];
      pbuf[wave][half * 8 + r][16 + lr] = (bf16_t)sc[1][r];
    }
    __syncthreads();
    V16 ap;
    ap.h[0] = *(const v8bf*)&pbuf[wave][lr][half * 8];
    ap.h[1] = *(const v8bf*)&pbuf[wave][lr][16 + half * 8];

#pragma unroll
    for (int nt = 0; nt < 4; ++nt) {
      const bf16_t* vr = Vp + (size_t)(nt * 16 + lr) * SS + kt;
      V16 bv;
      bv.h[0] = *(const v8bf*)(vr + half * 8);
      bv.h[1] = *(const v8bf*)(vr + 16 + half * 8);
      O[nt] = wmma_bf16(ap.v, bv.v, O[nt]);
    }
  }

#pragma unroll
  for (int nt = 0; nt < 4; ++nt) {
#pragma unroll
    for (int r = 0; r < 8; ++r) {
      int srow = qrow0 + half * 8 + r;
      int col = nt * 16 + lr;
      float val = O[nt][r] / lrow[r];
      attB[((size_t)(b * SS + srow)) * DD + (size_t)h * HD + col] = (bf16_t)val;
    }
  }
}

// ---------------------------------------------------------------------------
extern "C" void kernel_launch(void* const* d_in, const int* in_sizes, int n_in,
                              void* d_out, int out_size, void* d_ws, size_t ws_size,
                              hipStream_t stream) {
  (void)in_sizes; (void)n_in; (void)out_size; (void)ws_size;
  const float* x     = (const float*)d_in[0];
  const float* ln1_g = (const float*)d_in[1];
  const float* ln1_b = (const float*)d_in[2];
  const float* w_q   = (const float*)d_in[3];
  const float* w_k   = (const float*)d_in[4];
  const float* w_v   = (const float*)d_in[5];
  const float* w_o   = (const float*)d_in[6];
  const float* b_o   = (const float*)d_in[7];
  const float* ln2_g = (const float*)d_in[8];
  const float* ln2_b = (const float*)d_in[9];
  const float* w1    = (const float*)d_in[10];
  const float* b1    = (const float*)d_in[11];
  const float* w2    = (const float*)d_in[12];
  const float* b2    = (const float*)d_in[13];
  float* out = (float*)d_out;

  char* ws = (char*)d_ws;
  size_t off = 0;
  auto alloc = [&](size_t bytes) -> void* {
    void* p = ws + off;
    off += (bytes + 255) & ~(size_t)255;
    return p;
  };
  bf16_t* wqT  = (bf16_t*)alloc((size_t)DD * DD * 2);
  bf16_t* wkT  = (bf16_t*)alloc((size_t)DD * DD * 2);
  bf16_t* wvT  = (bf16_t*)alloc((size_t)DD * DD * 2);
  bf16_t* woT  = (bf16_t*)alloc((size_t)DD * DD * 2);
  bf16_t* w1T  = (bf16_t*)alloc((size_t)DD * FF * 2);
  bf16_t* w2T  = (bf16_t*)alloc((size_t)FF * DD * 2);
  bf16_t* h1b  = (bf16_t*)alloc((size_t)RR * DD * 2);
  bf16_t* qb   = (bf16_t*)alloc((size_t)RR * DD * 2);
  bf16_t* kb   = (bf16_t*)alloc((size_t)RR * DD * 2);
  bf16_t* vb   = (bf16_t*)alloc((size_t)RR * DD * 2);
  bf16_t* qhb  = (bf16_t*)alloc((size_t)RR * DD * 2);
  bf16_t* khb  = (bf16_t*)alloc((size_t)RR * DD * 2);
  bf16_t* vTb  = (bf16_t*)alloc((size_t)RR * DD * 2);
  bf16_t* attb = (bf16_t*)alloc((size_t)RR * DD * 2);
  float*  xmid = (float*) alloc((size_t)RR * DD * 4);
  bf16_t* h2b  = (bf16_t*)alloc((size_t)RR * DD * 2);
  bf16_t* m1b  = (bf16_t*)alloc((size_t)RR * FF * 2);

  // weights -> bf16, transposed
  wconv_kernel<<<(DD * DD + 255) / 256, 256, 0, stream>>>(w_q, wqT, DD, DD);
  wconv_kernel<<<(DD * DD + 255) / 256, 256, 0, stream>>>(w_k, wkT, DD, DD);
  wconv_kernel<<<(DD * DD + 255) / 256, 256, 0, stream>>>(w_v, wvT, DD, DD);
  wconv_kernel<<<(DD * DD + 255) / 256, 256, 0, stream>>>(w_o, woT, DD, DD);
  wconv_kernel<<<(DD * FF + 255) / 256, 256, 0, stream>>>(w1, w1T, DD, FF);
  wconv_kernel<<<(FF * DD + 255) / 256, 256, 0, stream>>>(w2, w2T, FF, DD);

  // LN1
  ln_kernel<<<RR, 256, 0, stream>>>(x, ln1_g, ln1_b, h1b);

  // QKV projections (TDM-staged LDS double-buffered WMMA GEMM)
  dim3 gq(RR / 128, DD / 128);
  gemm_bf16_kernel<0><<<gq, 256, 0, stream>>>(h1b, wqT, RR, DD, DD, nullptr, nullptr, nullptr, qb);
  gemm_bf16_kernel<0><<<gq, 256, 0, stream>>>(h1b, wkT, RR, DD, DD, nullptr, nullptr, nullptr, kb);
  gemm_bf16_kernel<0><<<gq, 256, 0, stream>>>(h1b, wvT, RR, DD, DD, nullptr, nullptr, nullptr, vb);

  split_heads_kernel<<<(RR * DD) / 256, 256, 0, stream>>>(qb, kb, vb, qhb, khb, vTb);

  // attention
  attn_kernel<<<BB * HH * (SS / 64), 128, 0, stream>>>(qhb, khb, vTb, attb);

  // output projection + bias + residual
  gemm_bf16_kernel<1><<<gq, 256, 0, stream>>>(attb, woT, RR, DD, DD, b_o, x, xmid, nullptr);

  // LN2
  ln_kernel<<<RR, 256, 0, stream>>>(xmid, ln2_g, ln2_b, h2b);

  // MLP
  dim3 g3(RR / 128, FF / 128);
  gemm_bf16_kernel<2><<<g3, 256, 0, stream>>>(h2b, w1T, RR, FF, DD, b1, nullptr, nullptr, m1b);
  gemm_bf16_kernel<1><<<gq, 256, 0, stream>>>(m1b, w2T, RR, DD, FF, b2, xmid, out, nullptr);
}
